// ImaginationModule_78039555768356
// MI455X (gfx1250) — matrix operand order repaired and checked by reading, not compiled, not run on hardware
//
#include <hip/hip_runtime.h>

typedef __attribute__((ext_vector_type(16))) __bf16        v16bf;
typedef __attribute__((ext_vector_type(8)))  float         v8f;
typedef __attribute__((ext_vector_type(4)))  unsigned int  u32x4;
typedef __attribute__((ext_vector_type(8)))  unsigned int  u32x8;

#define TM       64
#define XSTRIDE  72          // 64 cols + 8 pad (bf16 units)
#define HSTRIDE  520         // 512 cols + 8 pad (bf16 units)
#define XBYTES   (TM * XSTRIDE * 2)            // 9216 (per buffer, x2)
#define HBYTES   (TM * HSTRIDE * 2)            // 66560
#define SMEM_BYTES (2 * XBYTES + 2 * HBYTES + TM * 4)  // 151808

__device__ __forceinline__ unsigned short f2bf(float f) {
    unsigned int u = __builtin_bit_cast(unsigned int, f);
    u += 0x7FFFu + ((u >> 16) & 1u);           // round-to-nearest-even
    return (unsigned short)(u >> 16);
}

// A-fragment (16x32 bf16, MxK): lanes 0-15 row M=lane, K = k0..k0+15
//                               lanes 16-31 row M=lane-16, K = k0+16..k0+31
__device__ __forceinline__ v16bf load_a_frag(const unsigned short* base, int stride,
                                             int row0, int k0, int lane) {
    const int half = lane >> 4;
    const int r    = row0 + (lane & 15);
    const unsigned short* p = base + r * stride + k0 + half * 16;
    u32x4 lo = *(const u32x4*)(p);        // 8 bf16
    u32x4 hi = *(const u32x4*)(p + 8);    // 8 bf16, +8 elements
    u32x8 t;
    t[0] = lo[0]; t[1] = lo[1]; t[2] = lo[2]; t[3] = lo[3];
    t[4] = hi[0]; t[5] = hi[1]; t[6] = hi[2]; t[7] = hi[3];
    return __builtin_bit_cast(v16bf, t);
}

// B-fragment from per-lane base pointer + compile-time byte offset.
// bp already encodes (jbase + lane%16)*K + (lane/16)*16 ; byteOff = (jj*16*K + k0)*2
__device__ __forceinline__ v16bf load_b_imm(const unsigned short* bp, int byteOff) {
    const char* p = (const char*)bp + byteOff;
    u32x4 lo = *(const u32x4*)(p);
    u32x4 hi = *(const u32x4*)(p + 16);
    u32x8 t;
    t[0] = lo[0]; t[1] = lo[1]; t[2] = lo[2]; t[3] = lo[3];
    t[4] = hi[0]; t[5] = hi[1]; t[6] = hi[2]; t[7] = hi[3];
    return __builtin_bit_cast(v16bf, t);
}

// ---- prep: W[din,dout] f32 -> WT[dout,din] bf16 ----
__global__ void wt_transpose_bf16(const float* __restrict__ W,
                                  unsigned short* __restrict__ WT,
                                  int din, int dout, int n) {
    int i = blockIdx.x * 256 + threadIdx.x;
    if (i >= n) return;
    int o = i / din;
    int k = i - o * din;
    WT[i] = f2bf(W[(size_t)k * dout + o]);
}

// 8 x float2 strided loads with literal row stride (immediate offsets)
template <int SW>
__device__ __forceinline__ void ld8(const float* base, float2 st[8]) {
    #pragma unroll
    for (int e = 0; e < 8; ++e)
        st[e] = *(const float2*)(base + (size_t)e * 8 * SW);
}

// load one 64x64 chunk of the concatenated input into registers (8 x float2/thread)
__device__ __forceinline__ void stage_load(int cc, int tid, int rowBase,
                                           const float* src0, const float* text,
                                           const float* kw, const float* ctx,
                                           const float* spk, float2 st[8]) {
    const int cbase = cc * 64;
    const int r0 = (tid * 2) >> 6;     // 0..7
    const int c  = (tid * 2) & 63;
    if      (cbase < 512)  ld8<512>(src0 + (size_t)(rowBase + r0) * 512 + cbase          + c, st);
    else if (cbase < 1280) ld8<768>(text + (size_t)(rowBase + r0) * 768 + (cbase - 512)  + c, st);
    else if (cbase < 1536) ld8<256>(kw   + (size_t)(rowBase + r0) * 256 + (cbase - 1280) + c, st);
    else if (cbase < 1792) ld8<256>(ctx  + (size_t)(rowBase + r0) * 256 + (cbase - 1536) + c, st);
    else                   ld8<128>(spk  + (size_t)(rowBase + r0) * 128 + (cbase - 1792) + c, st);
}

// one 512-K GEMM layer: 4 M-tiles x 4 N-tiles per wave; 4 WMMAs per B fragment,
// B double-buffered with distance-2 prefetch (covers 8 WMMAs of latency)
__device__ __forceinline__ void gemm512(const unsigned short* Hsrc,
                                        const unsigned short* WT,
                                        int jbase, int lane,
                                        v8f acc[4][4]) {
    const int cl = lane & 15, half = lane >> 4;
    const unsigned short* bp = WT + (size_t)(jbase + cl) * 512 + half * 16;
    v16bf bq[2];
    bq[0] = load_b_imm(bp, 0);
    bq[1] = load_b_imm(bp, 16384);
    #pragma unroll 1
    for (int kt = 0; kt < 16; ++kt) {
        const int k0 = kt * 32;
        v16bf a[4];
        #pragma unroll
        for (int m = 0; m < 4; ++m)
            a[m] = load_a_frag(Hsrc, HSTRIDE, m * 16, k0, lane);
        #pragma unroll
        for (int jj = 0; jj < 4; ++jj) {
            // prefetch fragment (kt, jj+2), rolling into next kt (+64B) for jj>=2
            const int off = (((jj + 2) & 3) * 16384) + ((jj >= 2) ? 64 : 0);
            v16bf bn = load_b_imm(bp, off);
            #pragma unroll
            for (int m = 0; m < 4; ++m)
                acc[m][jj] = __builtin_amdgcn_wmma_f32_16x16x32_bf16(
                    false, a[m], false, bq[jj & 1], (short)0, acc[m][jj], false, false);
            bq[jj & 1] = bn;
        }
        bp += 32;                       // +64 bytes: next kt
    }
}

// ---- fused 3-layer MLP, one path per blockIdx.y ----
__global__ void __launch_bounds__(256) mlp_fused(
    const float* __restrict__ audio,  const float* __restrict__ visual,
    const float* __restrict__ text,   const float* __restrict__ kw,
    const float* __restrict__ ctx,    const float* __restrict__ spk,
    const int*   __restrict__ mask,
    const unsigned short* __restrict__ aW1T, const unsigned short* __restrict__ aW2T,
    const unsigned short* __restrict__ aW3T, const unsigned short* __restrict__ vW1T,
    const unsigned short* __restrict__ vW2T, const unsigned short* __restrict__ vW3T,
    const float* __restrict__ ab1, const float* __restrict__ ab2, const float* __restrict__ ab3,
    const float* __restrict__ vb1, const float* __restrict__ vb2, const float* __restrict__ vb3,
    float* __restrict__ out, int Btot)
{
    extern __shared__ char smem[];
    unsigned short* Xc0 = (unsigned short*)(smem);
    unsigned short* Xc1 = (unsigned short*)(smem + XBYTES);
    unsigned short* H1  = (unsigned short*)(smem + 2 * XBYTES);
    unsigned short* H2  = (unsigned short*)(smem + 2 * XBYTES + HBYTES);
    float*          FM  = (float*)(smem + 2 * XBYTES + 2 * HBYTES);

    const int tid     = threadIdx.x;
    const int lane    = tid & 31;
    const int wave    = tid >> 5;
    const int path    = blockIdx.y;
    const int rowBase = blockIdx.x * TM;

    const float* src0 = (path == 0) ? visual : audio;
    const unsigned short* W1T = (path == 0) ? aW1T : vW1T;
    const unsigned short* W2T = (path == 0) ? aW2T : vW2T;
    const unsigned short* W3T = (path == 0) ? aW3T : vW3T;
    const float* b1 = (path == 0) ? ab1 : vb1;
    const float* b2 = (path == 0) ? ab2 : vb2;
    const float* b3 = (path == 0) ? ab3 : vb3;
    float* outp = out + (size_t)path * (size_t)Btot * 512;

    if (tid < TM)
        FM[tid] = 1.0f - (float)mask[(size_t)(rowBase + tid) * 3 + path];

    const int jbase = wave * 64;             // each wave owns a unique 64-col slab
    const int rb    = (lane >> 4) * 8;       // C-frag row offset for this half-wave
    const int cl    = lane & 15;             // C-frag column lane

    v8f acc[4][4];
    #pragma unroll
    for (int m = 0; m < 4; ++m)
        #pragma unroll
        for (int j = 0; j < 4; ++j)
            #pragma unroll
            for (int v = 0; v < 8; ++v) acc[m][j][v] = 0.0f;

    // ---------------- layer 1: [64,1920] @ W1 -> H1 [64,512] ----------------
    float2 st[8];
    stage_load(0, tid, rowBase, src0, text, kw, ctx, spk, st);
    const unsigned short* bp1 = W1T + (size_t)(jbase + cl) * 1920 + (lane >> 4) * 16;
    v16bf bq[2];
    bq[0] = load_b_imm(bp1, 0);
    bq[1] = load_b_imm(bp1, 61440);
    #pragma unroll 1
    for (int cc = 0; cc < 30; ++cc) {
        unsigned short* Xb = (cc & 1) ? Xc1 : Xc0;
        // commit staged chunk cc (waits on its global loads here)
        #pragma unroll
        for (int e = 0; e < 8; ++e) {
            const int idx = (e * 256 + tid) * 2;
            const int r = idx >> 6, c = idx & 63;
            const unsigned int pk = (unsigned int)f2bf(st[e].x) |
                                    ((unsigned int)f2bf(st[e].y) << 16);
            *(unsigned int*)(Xb + r * XSTRIDE + c) = pk;
        }
        __syncthreads();
        // kick off global loads for chunk cc+1; they fly during the WMMAs below
        if (cc < 29)
            stage_load(cc + 1, tid, rowBase, src0, text, kw, ctx, spk, st);
        #pragma unroll
        for (int kt = 0; kt < 2; ++kt) {
            v16bf a[4];
            #pragma unroll
            for (int m = 0; m < 4; ++m)
                a[m] = load_a_frag(Xb, XSTRIDE, m * 16, kt * 32, lane);
            #pragma unroll
            for (int jj = 0; jj < 4; ++jj) {
                // prefetch fragment (kt, jj+2); jj>=2 rolls into next kt (+64B)
                const int off = kt * 64 + (((jj + 2) & 3) * 61440) + ((jj >= 2) ? 64 : 0);
                v16bf bn = load_b_imm(bp1, off);
                #pragma unroll
                for (int m = 0; m < 4; ++m)
                    acc[m][jj] = __builtin_amdgcn_wmma_f32_16x16x32_bf16(
                        false, a[m], false, bq[jj & 1], (short)0, acc[m][jj], false, false);
                bq[jj & 1] = bn;
            }
        }
        bp1 += 64;                      // +128 bytes: next chunk (2 k-tiles)
    }
    // bias + relu -> H1 (bf16)
    #pragma unroll
    for (int m = 0; m < 4; ++m)
        #pragma unroll
        for (int jj = 0; jj < 4; ++jj) {
            const int col = jbase + jj * 16 + cl;
            const float bv = b1[col];
            #pragma unroll
            for (int v = 0; v < 8; ++v) {
                float x = acc[m][jj][v] + bv;
                x = fmaxf(x, 0.0f);
                H1[(m * 16 + rb + v) * HSTRIDE + col] = f2bf(x);
            }
        }
    __syncthreads();

    // ---------------- layer 2: H1 @ W2 -> H2 ----------------
    #pragma unroll
    for (int m = 0; m < 4; ++m)
        #pragma unroll
        for (int j = 0; j < 4; ++j)
            #pragma unroll
            for (int v = 0; v < 8; ++v) acc[m][j][v] = 0.0f;
    gemm512(H1, W2T, jbase, lane, acc);
    #pragma unroll
    for (int m = 0; m < 4; ++m)
        #pragma unroll
        for (int jj = 0; jj < 4; ++jj) {
            const int col = jbase + jj * 16 + cl;
            const float bv = b2[col];
            #pragma unroll
            for (int v = 0; v < 8; ++v) {
                float x = acc[m][jj][v] + bv;
                x = fmaxf(x, 0.0f);
                H2[(m * 16 + rb + v) * HSTRIDE + col] = f2bf(x);
            }
        }
    __syncthreads();

    // ---------------- layer 3: H2 @ W3 -> out (masked) ----------------
    #pragma unroll
    for (int m = 0; m < 4; ++m)
        #pragma unroll
        for (int j = 0; j < 4; ++j)
            #pragma unroll
            for (int v = 0; v < 8; ++v) acc[m][j][v] = 0.0f;
    gemm512(H2, W3T, jbase, lane, acc);
    #pragma unroll
    for (int m = 0; m < 4; ++m)
        #pragma unroll
        for (int jj = 0; jj < 4; ++jj) {
            const int col = jbase + jj * 16 + cl;
            const float bv = b3[col];
            #pragma unroll
            for (int v = 0; v < 8; ++v) {
                const int lr = m * 16 + rb + v;
                const float x = (acc[m][jj][v] + bv) * FM[lr];
                outp[(size_t)(rowBase + lr) * 512 + col] = x;
            }
        }
}

extern "C" void kernel_launch(void* const* d_in, const int* in_sizes, int n_in,
                              void* d_out, int out_size, void* d_ws, size_t ws_size,
                              hipStream_t stream) {
    (void)n_in; (void)out_size; (void)ws_size;
    const float* audio  = (const float*)d_in[0];
    const float* visual = (const float*)d_in[1];
    const float* text   = (const float*)d_in[2];
    const float* kw     = (const float*)d_in[3];
    const float* ctx    = (const float*)d_in[4];
    const float* spk    = (const float*)d_in[5];
    const int*   mask   = (const int*)d_in[6];
    const float* aW1 = (const float*)d_in[7];  const float* ab1 = (const float*)d_in[8];
    const float* aW2 = (const float*)d_in[9];  const float* ab2 = (const float*)d_in[10];
    const float* aW3 = (const float*)d_in[11]; const float* ab3 = (const float*)d_in[12];
    const float* vW1 = (const float*)d_in[13]; const float* vb1 = (const float*)d_in[14];
    const float* vW2 = (const float*)d_in[15]; const float* vb2 = (const float*)d_in[16];
    const float* vW3 = (const float*)d_in[17]; const float* vb3 = (const float*)d_in[18];
    float* out = (float*)d_out;

    const int B = in_sizes[0] / 512;   // 32768

    // workspace layout (bf16 transposed weights); small tail slack absorbs dead prefetches
    unsigned short* ws   = (unsigned short*)d_ws;
    unsigned short* aW1T = ws;                       // 512*1920
    unsigned short* aW2T = aW1T + 512 * 1920;        // 512*512
    unsigned short* aW3T = aW2T + 512 * 512;
    unsigned short* vW1T = aW3T + 512 * 512;
    unsigned short* vW2T = vW1T + 512 * 1920;
    unsigned short* vW3T = vW2T + 512 * 512;

    {
        int n1 = 1920 * 512, n2 = 512 * 512;
        wt_transpose_bf16<<<(n1 + 255) / 256, 256, 0, stream>>>(aW1, aW1T, 1920, 512, n1);
        wt_transpose_bf16<<<(n2 + 255) / 256, 256, 0, stream>>>(aW2, aW2T, 512, 512, n2);
        wt_transpose_bf16<<<(n2 + 255) / 256, 256, 0, stream>>>(aW3, aW3T, 512, 512, n2);
        wt_transpose_bf16<<<(n1 + 255) / 256, 256, 0, stream>>>(vW1, vW1T, 1920, 512, n1);
        wt_transpose_bf16<<<(n2 + 255) / 256, 256, 0, stream>>>(vW2, vW2T, 512, 512, n2);
        wt_transpose_bf16<<<(n2 + 255) / 256, 256, 0, stream>>>(vW3, vW3T, 512, 512, n2);
    }

    dim3 grid(B / TM, 2);
    mlp_fused<<<grid, dim3(256), SMEM_BYTES, stream>>>(
        audio, visual, text, kw, ctx, spk, mask,
        aW1T, aW2T, aW3T, vW1T, vW2T, vW3T,
        ab1, ab2, ab3, vb1, vb2, vb3,
        out, B);
}
